// TrajectoryDecoder_14903536517879
// MI455X (gfx1250) — matrix operand order repaired and checked by reading, not compiled, not run on hardware
//
#include <hip/hip_runtime.h>

typedef unsigned short u16;
typedef __attribute__((ext_vector_type(16))) __bf16 v16bf;
typedef __attribute__((ext_vector_type(8)))  float  v8f;

#define B_TOT   4096
#define DM      128
#define H       256
#define G4      1024
#define PRED    45

union BfPack { uint4 u[2]; v16bf v; };

static __device__ __forceinline__ u16 f2bf(float x) {
    unsigned u = __float_as_uint(x);
    unsigned r = (u + 0x7FFFu + ((u >> 16) & 1u)) >> 16;
    return (u16)r;
}

// branch-free sigmoid / tanh on native v_exp_f32 (TRANS ops co-execute with WMMA)
static __device__ __forceinline__ float fast_sigmoid(float x) {
    return 1.0f / (1.0f + __expf(-x));
}
static __device__ __forceinline__ float fast_tanh(float x) {
    float e = __expf(2.0f * x);          // e->0 => -1 ; e->inf => +1 (no NaN path)
    return 1.0f - 2.0f / (e + 1.0f);
}

static __device__ __forceinline__ v8f wmma_bf16(v16bf a, v16bf b, v8f c) {
    return __builtin_amdgcn_wmma_f32_16x16x32_bf16(
        false, a, false, b, (short)0, c, false, false);
}

// A fragment (16x32 bf16) from LDS staging [16][256] bf16.
// A layout (contiguous, matches B): lanes 0-15 row=lane hold K {0..15}; lanes 16-31 row=lane-16 hold K {16..31}.
static __device__ __forceinline__ v16bf load_a_frag(const u16* sh, int kb, int lane) {
    int row  = lane & 15;
    int base = (lane & 16) ? 16 : 0;
    const u16* p = sh + row * 256 + kb * 32 + base;
    BfPack pk;
    pk.u[0] = *(const uint4*)(p);
    pk.u[1] = *(const uint4*)(p + 8);
    return pk.v;
}

// Four B fragments (one per gate) loaded with pinned global_load_b128 pairs.
// asm volatile => cannot be hoisted out of the 45-step loop (prevents the
// scratch-spill catastrophe); early-clobber outputs so in-flight loads never
// alias the address registers.
static __device__ __forceinline__ void gload_b4(const u16* p0, const u16* p1,
                                                const u16* p2, const u16* p3,
                                                BfPack& b0, BfPack& b1,
                                                BfPack& b2, BfPack& b3) {
    asm volatile(
        "global_load_b128 %0, %8, off\n\t"
        "global_load_b128 %1, %8, off offset:16\n\t"
        "global_load_b128 %2, %9, off\n\t"
        "global_load_b128 %3, %9, off offset:16\n\t"
        "global_load_b128 %4, %10, off\n\t"
        "global_load_b128 %5, %10, off offset:16\n\t"
        "global_load_b128 %6, %11, off\n\t"
        "global_load_b128 %7, %11, off offset:16\n\t"
        "s_wait_loadcnt 0x0"
        : "=&v"(b0.u[0]), "=&v"(b0.u[1]), "=&v"(b1.u[0]), "=&v"(b1.u[1]),
          "=&v"(b2.u[0]), "=&v"(b2.u[1]), "=&v"(b3.u[0]), "=&v"(b3.u[1])
        : "v"((unsigned long long)p0), "v"((unsigned long long)p1),
          "v"((unsigned long long)p2), "v"((unsigned long long)p3)
        : "memory");
}

static __device__ __forceinline__ void gload_b1(const u16* p, BfPack& b) {
    asm volatile(
        "global_load_b128 %0, %2, off\n\t"
        "global_load_b128 %1, %2, off offset:16\n\t"
        "s_wait_loadcnt 0x0"
        : "=&v"(b.u[0]), "=&v"(b.u[1])
        : "v"((unsigned long long)p)
        : "memory");
}

// ---------------- one-time init: h/c states + gates_ctx (fp32) ----------------
__global__ void init_state_kernel(const float* __restrict__ enc, const float* __restrict__ ctx,
                                  const float* __restrict__ Wh,  const float* __restrict__ bh,
                                  const float* __restrict__ Wc,  const float* __restrict__ bc,
                                  const float* __restrict__ Wih0,
                                  const float* __restrict__ bih0, const float* __restrict__ bhh0,
                                  float* __restrict__ h0w, float* __restrict__ h1w,
                                  float* __restrict__ c0w, float* __restrict__ c1w,
                                  float* __restrict__ gctx) {
    int idx = blockIdx.x * blockDim.x + threadIdx.x;
    if (idx >= B_TOT * 2048) return;
    int b = idx >> 11;
    int c = idx & 2047;
    if (c < 512) {
        const float* wr = Wh + c * DM;
        const float* e  = enc + b * DM;
        float s = bh[c];
        for (int k = 0; k < DM; ++k) s = fmaf(e[k], wr[k], s);
        if (c < 256) h0w[b * H + c] = s;
        else         h1w[b * H + (c - 256)] = s;
    } else if (c < 1024) {
        int cc = c - 512;
        const float* wr = Wc + cc * DM;
        const float* e  = enc + b * DM;
        float s = bc[cc];
        for (int k = 0; k < DM; ++k) s = fmaf(e[k], wr[k], s);
        if (cc < 256) c0w[b * H + cc] = s;
        else          c1w[b * H + (cc - 256)] = s;
    } else {
        int n = c - 1024;
        const float* wr = Wih0 + n * (2 + DM) + 2;   // skip the 2 pos columns
        const float* x  = ctx + b * DM;
        float s = bih0[n] + bhh0[n];
        for (int k = 0; k < DM; ++k) s = fmaf(x[k], wr[k], s);
        gctx[b * G4 + n] = s;
    }
}

// ---------------- weight fp32 -> bf16 conversion ----------------
__global__ void cvt_bf16_kernel(const float* __restrict__ s, u16* __restrict__ d, int n) {
    int i = blockIdx.x * blockDim.x + threadIdx.x;
    if (i < n) d[i] = f2bf(s[i]);
}

// ---------------- persistent 45-step decoder ----------------
// 256 blocks x 512 threads. Block owns 16 batch rows. Wave ns (0..15) owns
// units [ns*16, ns*16+16) across all 4 gates -> i/f/g/o of a unit share a
// register slot; cell state c lives in registers for all 45 steps.
__global__ __launch_bounds__(512, 1)
void decode_kernel(const float* __restrict__ gctx,
                   const float* __restrict__ h0w, const float* __restrict__ h1w,
                   const float* __restrict__ c0w, const float* __restrict__ c1w,
                   const u16* __restrict__ Whh0, const u16* __restrict__ Wih1,
                   const u16* __restrict__ Whh1, const u16* __restrict__ Wp1b,
                   const float* __restrict__ lastpos, const float* __restrict__ Wih0,
                   const float* __restrict__ bih1, const float* __restrict__ bhh1,
                   const float* __restrict__ bp1,  const float* __restrict__ Wp2,
                   const float* __restrict__ bp2,  float* __restrict__ out) {
    __shared__ u16   sh_h0[16 * 256];     // bf16 A staging, h0
    __shared__ u16   sh_h1[16 * 256];     // bf16 A staging, h1
    __shared__ float sh_hidden[16 * 64];  // relu(h1@Wp1^T+bp1)
    __shared__ float sh_pos[32];          // 16 x 2
    __shared__ float sh_Wp2[128];         // [2][64]
    __shared__ float sh_bp2[2];

    const int tid  = threadIdx.x;
    const int lane = tid & 31;
    const int ns   = tid >> 5;             // wave id == unit slice
    const int rowBase  = blockIdx.x * 16;
    const int mrow     = (lane & 16) ? 8 : 0;   // C/D layout row offset
    const int ncol     = lane & 15;             // C/D layout column
    const int lane_row = ns * 16 + ncol;        // weight row within a gate block
    const int koff     = (lane & 16) ? 16 : 0;  // B layout K offset

    // ---- LDS init ----
    for (int i = tid; i < 4096; i += 512) {
        int m = i >> 8, k = i & 255;
        sh_h0[i] = f2bf(h0w[(rowBase + m) * H + k]);
        sh_h1[i] = f2bf(h1w[(rowBase + m) * H + k]);
    }
    if (tid < 32)  sh_pos[tid] = lastpos[rowBase * 2 + tid];
    if (tid < 128) sh_Wp2[tid] = Wp2[tid];
    if (tid < 2)   sh_bp2[tid] = bp2[tid];

    // ---- register-resident loop invariants ----
    float c0f[8], c1f[8];          // cell states (C/D layout)
    float gcache[4][8];            // gates_ctx slice
    float wx[4], wy[4], bc1[4];    // W_pos columns, fused layer-1 bias
#pragma unroll
    for (int r = 0; r < 8; ++r) {
        int m = rowBase + r + mrow;
        c0f[r] = c0w[m * H + lane_row];
        c1f[r] = c1w[m * H + lane_row];
    }
#pragma unroll
    for (int g = 0; g < 4; ++g) {
        int n = g * 256 + lane_row;
        wx[g]  = Wih0[n * (2 + DM) + 0];
        wy[g]  = Wih0[n * (2 + DM) + 1];
        bc1[g] = bih1[n] + bhh1[n];
#pragma unroll
        for (int r = 0; r < 8; ++r)
            gcache[g][r] = gctx[(rowBase + r + mrow) * G4 + n];
    }
    const float bp1v = (ns < 4) ? bp1[lane_row] : 0.0f;

    // per-lane weight base pointers (row-major [1024][256] bf16; gate stride 256*256)
    const u16* W0base = Whh0 + lane_row * H + koff;
    const u16* W1base = Wih1 + lane_row * H + koff;
    const u16* W2base = Whh1 + lane_row * H + koff;
    const u16* Wpbase = Wp1b + lane_row * H + koff;   // valid for ns<4

    __syncthreads();

    for (int t = 0; t < PRED; ++t) {
        v8f acc[4];

        // ---- layer-0 acc init: gates_ctx + pos @ W_pos^T ----
        float px[8], py[8];
#pragma unroll
        for (int r = 0; r < 8; ++r) {
            int m = r + mrow;
            px[r] = sh_pos[m * 2];
            py[r] = sh_pos[m * 2 + 1];
        }
#pragma unroll
        for (int g = 0; g < 4; ++g)
#pragma unroll
            for (int r = 0; r < 8; ++r)
                acc[g][r] = fmaf(py[r], wy[g], fmaf(px[r], wx[g], gcache[g][r]));

        // ---- GEMM0: += h0 @ W_hh0^T ----
        for (int kb = 0; kb < 8; ++kb) {
            v16bf aF = load_a_frag(sh_h0, kb, lane);
            BfPack b0, b1, b2, b3;
            const u16* p = W0base + kb * 32;
            gload_b4(p, p + 65536, p + 131072, p + 196608, b0, b1, b2, b3);
            acc[0] = wmma_bf16(aF, b0.v, acc[0]);
            acc[1] = wmma_bf16(aF, b1.v, acc[1]);
            acc[2] = wmma_bf16(aF, b2.v, acc[2]);
            acc[3] = wmma_bf16(aF, b3.v, acc[3]);
        }
        __syncthreads();
        // ---- LSTM cell 0 (registers only, branch-free activations) ----
#pragma unroll
        for (int r = 0; r < 8; ++r) {
            float c = fast_sigmoid(acc[1][r]) * c0f[r]
                    + fast_sigmoid(acc[0][r]) * fast_tanh(acc[2][r]);
            float h = fast_sigmoid(acc[3][r]) * fast_tanh(c);
            c0f[r] = c;
            sh_h0[(r + mrow) * 256 + lane_row] = f2bf(h);
        }
        __syncthreads();

        // ---- layer-1: bias1 + h0 @ W_ih1^T + h1 @ W_hh1^T ----
#pragma unroll
        for (int g = 0; g < 4; ++g)
#pragma unroll
            for (int r = 0; r < 8; ++r) acc[g][r] = bc1[g];
        for (int kb = 0; kb < 8; ++kb) {
            v16bf aF = load_a_frag(sh_h0, kb, lane);
            BfPack b0, b1, b2, b3;
            const u16* p = W1base + kb * 32;
            gload_b4(p, p + 65536, p + 131072, p + 196608, b0, b1, b2, b3);
            acc[0] = wmma_bf16(aF, b0.v, acc[0]);
            acc[1] = wmma_bf16(aF, b1.v, acc[1]);
            acc[2] = wmma_bf16(aF, b2.v, acc[2]);
            acc[3] = wmma_bf16(aF, b3.v, acc[3]);
        }
        for (int kb = 0; kb < 8; ++kb) {
            v16bf aF = load_a_frag(sh_h1, kb, lane);
            BfPack b0, b1, b2, b3;
            const u16* p = W2base + kb * 32;
            gload_b4(p, p + 65536, p + 131072, p + 196608, b0, b1, b2, b3);
            acc[0] = wmma_bf16(aF, b0.v, acc[0]);
            acc[1] = wmma_bf16(aF, b1.v, acc[1]);
            acc[2] = wmma_bf16(aF, b2.v, acc[2]);
            acc[3] = wmma_bf16(aF, b3.v, acc[3]);
        }
        __syncthreads();
        // ---- LSTM cell 1 ----
#pragma unroll
        for (int r = 0; r < 8; ++r) {
            float c = fast_sigmoid(acc[1][r]) * c1f[r]
                    + fast_sigmoid(acc[0][r]) * fast_tanh(acc[2][r]);
            float h = fast_sigmoid(acc[3][r]) * fast_tanh(c);
            c1f[r] = c;
            sh_h1[(r + mrow) * 256 + lane_row] = f2bf(h);
        }
        __syncthreads();

        // ---- projection: hidden = relu(h1 @ Wp1^T + bp1); waves ns<4, one 16x16 tile each ----
        if (ns < 4) {
            v8f accp;
#pragma unroll
            for (int r = 0; r < 8; ++r) accp[r] = bp1v;
            for (int kb = 0; kb < 8; ++kb) {
                v16bf aF = load_a_frag(sh_h1, kb, lane);
                BfPack bp;
                gload_b1(Wpbase + kb * 32, bp);
                accp = wmma_bf16(aF, bp.v, accp);
            }
#pragma unroll
            for (int r = 0; r < 8; ++r) {
                float v = accp[r];
                sh_hidden[(r + mrow) * 64 + lane_row] = v > 0.0f ? v : 0.0f;
            }
        }
        __syncthreads();

        // ---- delta = hidden @ Wp2^T + bp2 ; pos += delta ; emit traj ----
        if (tid < 32) {
            int m = tid >> 1, d = tid & 1;
            float s = sh_bp2[d];
#pragma unroll
            for (int k = 0; k < 64; ++k)
                s = fmaf(sh_hidden[m * 64 + k], sh_Wp2[d * 64 + k], s);
            float p = sh_pos[tid] + s;
            out[(rowBase + m) * (PRED * 2) + t * 2 + d] = p;
            sh_pos[tid] = p;
        }
        __syncthreads();
    }
}

extern "C" void kernel_launch(void* const* d_in, const int* in_sizes, int n_in,
                              void* d_out, int out_size, void* d_ws, size_t ws_size,
                              hipStream_t stream) {
    const float* enc     = (const float*)d_in[0];
    const float* lastpos = (const float*)d_in[1];
    const float* ctx     = (const float*)d_in[2];
    const float* Wh      = (const float*)d_in[3];
    const float* bh      = (const float*)d_in[4];
    const float* Wc      = (const float*)d_in[5];
    const float* bc      = (const float*)d_in[6];
    const float* Wih0    = (const float*)d_in[7];
    const float* Whh0    = (const float*)d_in[8];
    const float* bih0    = (const float*)d_in[9];
    const float* bhh0    = (const float*)d_in[10];
    const float* Wih1    = (const float*)d_in[11];
    const float* Whh1    = (const float*)d_in[12];
    const float* bih1    = (const float*)d_in[13];
    const float* bhh1    = (const float*)d_in[14];
    const float* Wp1     = (const float*)d_in[15];
    const float* bp1     = (const float*)d_in[16];
    const float* Wp2     = (const float*)d_in[17];
    const float* bp2     = (const float*)d_in[18];
    float* out = (float*)d_out;

    // workspace carve-up
    float* h0w  = (float*)d_ws;              // B*H
    float* h1w  = h0w + B_TOT * H;
    float* c0w  = h1w + B_TOT * H;
    float* c1w  = c0w + B_TOT * H;
    float* gctx = c1w + B_TOT * H;           // B*G4
    u16* Whh0b = (u16*)(gctx + B_TOT * G4);  // G4*H
    u16* Wih1b = Whh0b + G4 * H;
    u16* Whh1b = Wih1b + G4 * H;
    u16* Wp1b  = Whh1b + G4 * H;             // 64*H

    // 1) init states + gates_ctx
    {
        int total = B_TOT * 2048;
        init_state_kernel<<<(total + 255) / 256, 256, 0, stream>>>(
            enc, ctx, Wh, bh, Wc, bc, Wih0, bih0, bhh0,
            h0w, h1w, c0w, c1w, gctx);
    }
    // 2) convert recurrent weights to bf16
    cvt_bf16_kernel<<<(G4 * H + 255) / 256, 256, 0, stream>>>(Whh0, Whh0b, G4 * H);
    cvt_bf16_kernel<<<(G4 * H + 255) / 256, 256, 0, stream>>>(Wih1, Wih1b, G4 * H);
    cvt_bf16_kernel<<<(G4 * H + 255) / 256, 256, 0, stream>>>(Whh1, Whh1b, G4 * H);
    cvt_bf16_kernel<<<(64 * H + 255) / 256, 256, 0, stream>>>(Wp1, Wp1b, 64 * H);

    // 3) persistent 45-step decode: 256 blocks x 512 threads (16 batch rows / block)
    decode_kernel<<<B_TOT / 16, 512, 0, stream>>>(
        gctx, h0w, h1w, c0w, c1w,
        Whh0b, Wih1b, Whh1b, Wp1b,
        lastpos, Wih0, bih1, bhh1, bp1, Wp2, bp2, out);
}
